// GINE_55843164783469
// MI455X (gfx1250) — compile-verified, matrix operand order checked by reading, not run on hardware
//
#include <hip/hip_runtime.h>
#include <hip/hip_bf16.h>

// GINE on MI455X (gfx1250, wave32). Strategy:
//  - Per layer: edge kernel (fused edge_attr@We GEMM via v_wmma_f32_16x16x32_f16,
//    + h[src] gather + relu + atomic scatter-add into L2-resident agg),
//    then node-MLP kernel (two 64x64 GEMMs via WMMA), then final mean kernel.
//  - HBM roofline ~220 MB/layer (edge_attr stream dominates); h/agg (25.6 MB each)
//    live in the 192 MB L2, so gather+atomics are L2 traffic.
//  - f16 WMMA keeps the 6.5 GFLOP/layer edge GEMM on the matrix pipe (f32 accum).
//  - Scatter phase batches 4 edges (8 h-loads + 8 LDS reads in flight) before the
//    dependent relu+atomic chain, and uses readfirstlane so uniform row addresses
//    are computed on the SALU.

typedef __attribute__((ext_vector_type(16))) _Float16 v16h;
typedef __attribute__((ext_vector_type(8)))  float    v8f;

#define WAVES_PER_BLOCK 8
#define GINE_D   64
#define GINE_ED  32

// K index covered by half `hi` (0..15) of a v16h A/B fragment for 16x16x32 f16 WMMA.
// Lane half selects koff: lanes 0-15 -> koff=0, lanes 16-31 -> koff=8.
// Pattern (ISA 7.12.2, 16-bit A 16x32): V0-3 hold K {0..7 | 8..15}, V4-7 hold K {16..23 | 24..31}.
__device__ __forceinline__ int frag_k(int hi, int koff) {
    return ((hi < 8) ? hi : hi + 8) + koff;
}

// ---------------------------------------------------------------------------
// Edge kernel: for each 16-edge tile (one wave):
//   lin = edge_attr[tile] @ We + be          (WMMA, f32 accum seeded with be)
//   msg = relu(h[src] + lin)                 (via LDS staging, coalesced)
//   agg[dst] += msg                          (global_atomic_add_f32, coalesced)
// ---------------------------------------------------------------------------
__global__ __launch_bounds__(256) void gine_edge_kernel(
    const float* __restrict__ h,
    const float* __restrict__ edge_attr,
    const float* __restrict__ We,   // [32 x 64] row-major (layer slice)
    const float* __restrict__ be,   // [64]
    const int*   __restrict__ src,
    const int*   __restrict__ dst,
    float*       __restrict__ agg,
    int n_edges)
{
    __shared__ float lds_msg[WAVES_PER_BLOCK][16][GINE_D];

    const int tid  = threadIdx.x;
    const int wave = tid >> 5;
    const int lane = tid & 31;
    const int ln   = lane & 15;
    const int koff = (lane < 16) ? 0 : 8;
    const int mbase = (lane < 16) ? 0 : 8;

    // B fragments of We for the 4 output 16-col tiles; built once, kept in VGPRs.
    v16h bfrag[4];
    float bias[4];
#pragma unroll
    for (int t = 0; t < 4; ++t) {
        const int n = t * 16 + ln;
#pragma unroll
        for (int hi = 0; hi < 16; ++hi) {
            bfrag[t][hi] = (_Float16)We[frag_k(hi, koff) * GINE_D + n];
        }
        bias[t] = be[n];
    }

    const int n_tiles    = n_edges >> 4;                    // 16 edges per tile
    const int tile_step  = gridDim.x * WAVES_PER_BLOCK;

    for (int tile = blockIdx.x * WAVES_PER_BLOCK + wave; tile < n_tiles; tile += tile_step) {
        const int e0   = tile << 4;
        const int erow = e0 + ln;

        // A fragment: 16 edge rows of edge_attr (f32 -> f16).
        const float* ea = edge_attr + (size_t)erow * GINE_ED;
        // Prefetch the next tile's edge_attr stream (speculative, gfx1250 global_prefetch_b8).
        __builtin_prefetch(ea + (size_t)tile_step * 16 * GINE_ED, 0, 0);
        v16h a;
#pragma unroll
        for (int hi = 0; hi < 16; ++hi) {
            a[hi] = (_Float16)ea[frag_k(hi, koff)];
        }

#pragma unroll
        for (int t = 0; t < 4; ++t) {
            v8f c;
#pragma unroll
            for (int r = 0; r < 8; ++r) c[r] = bias[t];
            c = __builtin_amdgcn_wmma_f32_16x16x32_f16(
                    false, a, false, bfrag[t], (short)0, c, false, false);
            // C/D layout: lane holds column N = t*16+ln; VGPR r -> row M = mbase + r.
#pragma unroll
            for (int r = 0; r < 8; ++r)
                lds_msg[wave][mbase + r][t * 16 + ln] = c[r];
        }

        // Scatter phase, 4 edges per batch: issue all 8 h-loads + 8 LDS reads
        // before the dependent add/relu/atomic chain (MLP against L2 latency).
#pragma unroll
        for (int mq = 0; mq < 16; mq += 4) {
            float hv[8], mv[8];
            int   dvid[4];
#pragma unroll
            for (int j = 0; j < 4; ++j) {
                const int e = e0 + mq + j;                       // wave-uniform
                const int s = __builtin_amdgcn_readfirstlane(src[e]);
                dvid[j]     = __builtin_amdgcn_readfirstlane(dst[e]);
                const float* hs = h + (size_t)s * GINE_D;
                hv[2 * j]     = hs[lane];
                hv[2 * j + 1] = hs[lane + 32];
                mv[2 * j]     = lds_msg[wave][mq + j][lane];
                mv[2 * j + 1] = lds_msg[wave][mq + j][lane + 32];
            }
#pragma unroll
            for (int j = 0; j < 4; ++j) {
                float* ag = agg + (size_t)dvid[j] * GINE_D;
                atomicAdd(&ag[lane],      fmaxf(hv[2 * j]     + mv[2 * j],     0.0f));
                atomicAdd(&ag[lane + 32], fmaxf(hv[2 * j + 1] + mv[2 * j + 1], 0.0f));
            }
        }
    }
}

// ---------------------------------------------------------------------------
// Node kernel: z = h + agg; z1 = relu(z@W1 + b1); out = z1@W2 + b2 (opt relu)
// One wave per 16-node tile; both 64x64 GEMMs on WMMA (K=32 x 2 steps).
// ---------------------------------------------------------------------------
__global__ __launch_bounds__(256) void gine_node_kernel(
    const float* __restrict__ h,
    const float* __restrict__ agg,
    const float* __restrict__ W1,   // [64 x 64] row-major (layer slice)
    const float* __restrict__ b1,
    const float* __restrict__ W2,
    const float* __restrict__ b2,
    float*       __restrict__ h_out,
    int n_nodes, int apply_relu)
{
    __shared__ float lds_z[WAVES_PER_BLOCK][16][GINE_D];

    const int tid  = threadIdx.x;
    const int wave = tid >> 5;
    const int lane = tid & 31;
    const int ln   = lane & 15;
    const int koff = (lane < 16) ? 0 : 8;
    const int mbase = (lane < 16) ? 0 : 8;

    // Weight B-fragments: [N-tile][K-frag], kept in VGPRs for the whole kernel.
    v16h w1f[4][2], w2f[4][2];
    float bias1[4], bias2[4];
#pragma unroll
    for (int t = 0; t < 4; ++t) {
        const int n = t * 16 + ln;
#pragma unroll
        for (int f = 0; f < 2; ++f) {
#pragma unroll
            for (int hi = 0; hi < 16; ++hi) {
                const int k = f * 32 + frag_k(hi, koff);
                w1f[t][f][hi] = (_Float16)W1[k * GINE_D + n];
                w2f[t][f][hi] = (_Float16)W2[k * GINE_D + n];
            }
        }
        bias1[t] = b1[n];
        bias2[t] = b2[n];
    }

    const int n_tiles   = n_nodes >> 4;                 // 16 nodes per tile
    const int tile_step = gridDim.x * WAVES_PER_BLOCK;

    for (int tile = blockIdx.x * WAVES_PER_BLOCK + wave; tile < n_tiles; tile += tile_step) {
        const int row = tile * 16 + ln;                 // A-matrix row M = ln
        const float* hr = h   + (size_t)row * GINE_D;
        const float* ar = agg + (size_t)row * GINE_D;

        // A fragments of z = h + agg (two K=32 fragments).
        v16h a0, a1;
#pragma unroll
        for (int hi = 0; hi < 16; ++hi) {
            const int k = frag_k(hi, koff);
            a0[hi] = (_Float16)(hr[k]      + ar[k]);
            a1[hi] = (_Float16)(hr[32 + k] + ar[32 + k]);
        }

        // GEMM1 + relu -> LDS
#pragma unroll
        for (int t = 0; t < 4; ++t) {
            v8f c;
#pragma unroll
            for (int r = 0; r < 8; ++r) c[r] = bias1[t];
            c = __builtin_amdgcn_wmma_f32_16x16x32_f16(false, a0, false, w1f[t][0], (short)0, c, false, false);
            c = __builtin_amdgcn_wmma_f32_16x16x32_f16(false, a1, false, w1f[t][1], (short)0, c, false, false);
#pragma unroll
            for (int r = 0; r < 8; ++r)
                lds_z[wave][mbase + r][t * 16 + ln] = fmaxf(c[r], 0.0f);
        }

        // Rebuild A fragments from LDS (same-wave LDS ordering; waits inserted by compiler).
        v16h z0, z1;
#pragma unroll
        for (int hi = 0; hi < 16; ++hi) {
            const int k = frag_k(hi, koff);
            z0[hi] = (_Float16)lds_z[wave][ln][k];
            z1[hi] = (_Float16)lds_z[wave][ln][32 + k];
        }

        // GEMM2 (+ optional inter-layer relu) -> LDS -> coalesced global store.
#pragma unroll
        for (int t = 0; t < 4; ++t) {
            v8f c;
#pragma unroll
            for (int r = 0; r < 8; ++r) c[r] = bias2[t];
            c = __builtin_amdgcn_wmma_f32_16x16x32_f16(false, z0, false, w2f[t][0], (short)0, c, false, false);
            c = __builtin_amdgcn_wmma_f32_16x16x32_f16(false, z1, false, w2f[t][1], (short)0, c, false, false);
#pragma unroll
            for (int r = 0; r < 8; ++r) {
                float v = c[r];
                if (apply_relu) v = fmaxf(v, 0.0f);
                lds_z[wave][mbase + r][t * 16 + ln] = v;
            }
        }
#pragma unroll
        for (int m = 0; m < 16; ++m) {
            float* out = h_out + (size_t)(tile * 16 + m) * GINE_D;
            out[lane]      = lds_z[wave][m][lane];
            out[lane + 32] = lds_z[wave][m][lane + 32];
        }
    }
}

// ---------------------------------------------------------------------------
// Mean over nodes -> out[64] (d_out pre-zeroed). Scaled atomics, LDS reduce.
// ---------------------------------------------------------------------------
__global__ __launch_bounds__(256) void gine_mean_kernel(
    const float* __restrict__ h, float* __restrict__ out, int n_nodes)
{
    __shared__ float red[256];
    const int tid = threadIdx.x;
    const int dim = tid & 63;
    const int r   = tid >> 6;     // 4 node-rows per block pass

    float acc = 0.0f;
    for (int node = blockIdx.x * 4 + r; node < n_nodes; node += gridDim.x * 4)
        acc += h[(size_t)node * GINE_D + dim];
    red[tid] = acc;
    __syncthreads();
    if (tid < 64) {
        float s = red[tid] + red[tid + 64] + red[tid + 128] + red[tid + 192];
        atomicAdd(&out[tid], s * (1.0f / (float)n_nodes));
    }
}

// ---------------------------------------------------------------------------
extern "C" void kernel_launch(void* const* d_in, const int* in_sizes, int n_in,
                              void* d_out, int out_size, void* d_ws, size_t ws_size,
                              hipStream_t stream) {
    const float* x         = (const float*)d_in[0];   // [N, 64]
    const float* edge_attr = (const float*)d_in[1];   // [E, 32]
    const float* We        = (const float*)d_in[2];   // [L, 32, 64]
    const float* be        = (const float*)d_in[3];   // [L, 64]
    const float* W1        = (const float*)d_in[4];   // [L, 64, 64]
    const float* b1        = (const float*)d_in[5];   // [L, 64]
    const float* W2        = (const float*)d_in[6];   // [L, 64, 64]
    const float* b2        = (const float*)d_in[7];   // [L, 64]
    const int*   ei        = (const int*)d_in[8];     // [2, E]

    const int n_nodes  = in_sizes[0] / GINE_D;
    const int n_edges  = in_sizes[8] / 2;
    const int n_layers = in_sizes[2] / (GINE_ED * GINE_D);

    const int* src = ei;
    const int* dst = ei + n_edges;

    // Workspace: hA | hB | agg  (each N*64 floats).
    float* hA  = (float*)d_ws;
    float* hB  = hA + (size_t)n_nodes * GINE_D;
    float* agg = hB + (size_t)n_nodes * GINE_D;

    const int edge_tiles  = n_edges >> 4;
    const int node_tiles  = n_nodes >> 4;
    const int edge_blocks = (edge_tiles + WAVES_PER_BLOCK - 1) / WAVES_PER_BLOCK;
    const int node_blocks = (node_tiles + WAVES_PER_BLOCK - 1) / WAVES_PER_BLOCK;

    const float* h_cur = x;
    float* bufs[2] = { hA, hB };

    for (int l = 0; l < n_layers; ++l) {
        hipMemsetAsync(agg, 0, (size_t)n_nodes * GINE_D * sizeof(float), stream);

        gine_edge_kernel<<<edge_blocks, 256, 0, stream>>>(
            h_cur, edge_attr,
            We + (size_t)l * GINE_ED * GINE_D, be + (size_t)l * GINE_D,
            src, dst, agg, n_edges);

        float* h_next = bufs[l & 1];
        gine_node_kernel<<<node_blocks, 256, 0, stream>>>(
            h_cur, agg,
            W1 + (size_t)l * GINE_D * GINE_D, b1 + (size_t)l * GINE_D,
            W2 + (size_t)l * GINE_D * GINE_D, b2 + (size_t)l * GINE_D,
            h_next, n_nodes, (l < n_layers - 1) ? 1 : 0);

        h_cur = h_next;
    }

    hipMemsetAsync(d_out, 0, (size_t)out_size * sizeof(float), stream);
    gine_mean_kernel<<<256, 256, 0, stream>>>(h_cur, (float*)d_out, n_nodes);
}
